// SimpleSNN_24275155157111
// MI455X (gfx1250) — matrix operand order repaired and checked by reading, not compiled, not run on hardware
//
#include <hip/hip_runtime.h>

typedef __attribute__((ext_vector_type(16))) _Float16 v16h;
typedef __attribute__((ext_vector_type(8)))  _Float16 v8h;
typedef __attribute__((ext_vector_type(8)))  float    v8f;
typedef __attribute__((ext_vector_type(4)))  float    v4f;

#define IN_DIM   128
#define HIDDEN   512
#define NCLASS   10
#define T_STEPS  512
#define BATCH    512
#define BETA     0.9f
#define THRESH   0.15f

#define BTILE    16
#define HCHUNK   64
#define NTHREADS 128
#define XPITCH   136          // 128 halfs + pad; 272B rows keep 16B alignment
#define WPITCH   136
#define SPITCH   (HCHUNK + 4)

__global__ void snn_init_out(const float* __restrict__ b2, float* __restrict__ out) {
  int i = blockIdx.x * blockDim.x + threadIdx.x;
  if (i < BATCH * NCLASS) out[i] = b2[i % NCLASS];   // acc/T = partials/T + b2
}

__global__ __launch_bounds__(NTHREADS) void snn_fused(
    const float* __restrict__ x, const float* __restrict__ W1,
    const float* __restrict__ b1, const float* __restrict__ W2,
    float* __restrict__ out)
{
  __shared__ __align__(16) _Float16 sW[HCHUNK][WPITCH];
  __shared__ __align__(16) _Float16 sX[2][BTILE][XPITCH];   // 2 timesteps per barrier pair
  __shared__ __align__(16) float    sS[BTILE][SPITCH];

  const int tid   = threadIdx.x;
  const int lane  = tid & 31;
  const int wave  = tid >> 5;
  const int hbase = blockIdx.x * HCHUNK;
  const int bbase = blockIdx.y * BTILE;

  // ---- Stage 2*W1 chunk (64 h-rows x 128 d-cols) into LDS as f16, once ----
  // (the reference's 2*x scaling is folded into W1: done 1x here instead of 512x on x)
  {
    const int row = tid >> 1;
    const int col = (tid & 1) * 64;
    const float* src = W1 + (size_t)(hbase + row) * IN_DIM + col;
    _Float16* dst = &sW[row][col];
#pragma unroll
    for (int i = 0; i < 8; ++i) {
      v4f a = *(const v4f*)(src + i * 8);
      v4f b = *(const v4f*)(src + i * 8 + 4);
      v8h h;
#pragma unroll
      for (int j = 0; j < 4; ++j) {
        h[j]     = (_Float16)(2.0f * a[j]);
        h[4 + j] = (_Float16)(2.0f * b[j]);
      }
      *(v8h*)(dst + i * 8) = h;
    }
  }
  __syncthreads();

  // ---- Preload this wave's B fragments: (2*W1)^T[128 x 16] as 4 x (32x16) ----
  // B as loaded here (k order matches A): N = lane%16; lane<16 holds K=0..15, lane>=16 K=16..31.
  v16h bfrag[4];
  {
    const int n  = lane & 15;
    const int kb = (lane >> 4) * 16;
    const int hh = wave * 16 + n;
#pragma unroll
    for (int kk = 0; kk < 4; ++kk) {
      v8h lo = *(const v8h*)&sW[hh][kk * 32 + kb];
      v8h hi = *(const v8h*)&sW[hh][kk * 32 + kb + 8];
#pragma unroll
      for (int j = 0; j < 8; ++j) { bfrag[kk][j] = lo[j]; bfrag[kk][8 + j] = hi[j]; }
    }
  }

  // b1 folded into acc0's C operand (same value for all 8 C elements of a lane)
  const float b1v = b1[hbase + wave * 16 + (lane & 15)];
  v8f binit;
#pragma unroll
  for (int r = 0; r < 8; ++r) binit[r] = b1v;

  // A as loaded here (k order matches B): M = lane%16; lane<16 holds K 0..15, lane>=16 K 16..31
  const int arow = lane & 15;
  const int akof = (lane >> 4) * 16;

  // x staging: 8 threads per b-row, 16 floats (d) each
  const int xrow = tid >> 3;
  const int xcol = (tid & 7) * 16;
  const float* xsrc0 = x + ((size_t)(bbase + xrow) * T_STEPS) * IN_DIM + xcol;

  float memv[8], cnt[8], spk[8];
#pragma unroll
  for (int r = 0; r < 8; ++r) { memv[r] = 0.0f; cnt[r] = 0.0f; spk[r] = 0.0f; }

  for (int t0 = 0; t0 < T_STEPS; t0 += 2) {
    // issue both timesteps' x loads before the barrier -> overlap with prior compute
    v4f rg[8];
#pragma unroll
    for (int u = 0; u < 2; ++u) {
      const float* xs = xsrc0 + (size_t)(t0 + u) * IN_DIM;
#pragma unroll
      for (int q = 0; q < 4; ++q) rg[u * 4 + q] = *(const v4f*)(xs + q * 4);
    }

    __syncthreads();                 // previous pair done reading sX
#pragma unroll
    for (int u = 0; u < 2; ++u) {
      v8h h0, h1;
#pragma unroll
      for (int j = 0; j < 4; ++j) {
        h0[j]     = (_Float16)rg[u * 4 + 0][j];
        h0[4 + j] = (_Float16)rg[u * 4 + 1][j];
        h1[j]     = (_Float16)rg[u * 4 + 2][j];
        h1[4 + j] = (_Float16)rg[u * 4 + 3][j];
      }
      *(v8h*)&sX[u][xrow][xcol]     = h0;
      *(v8h*)&sX[u][xrow][xcol + 8] = h1;
    }
    __syncthreads();                 // sX(t0), sX(t0+1) ready

#pragma unroll
    for (int u = 0; u < 2; ++u) {
      // preload all 4 A fragments -> one dscnt wait, then 4 back-to-back WMMAs
      v16h a[4];
#pragma unroll
      for (int kk = 0; kk < 4; ++kk) {
        v8h lo = *(const v8h*)&sX[u][arow][kk * 32 + akof];
        v8h hi = *(const v8h*)&sX[u][arow][kk * 32 + akof + 8];
#pragma unroll
        for (int j = 0; j < 8; ++j) { a[kk][j] = lo[j]; a[kk][8 + j] = hi[j]; }
      }

      // two independent accumulation chains (break 4-deep WMMA RAW chain)
      v8f acc0 = binit;              // carries the +b1 term
      v8f acc1 = {};
      acc0 = __builtin_amdgcn_wmma_f32_16x16x32_f16(false, a[0], false, bfrag[0],
                                                    (short)0, acc0, false, false);
      acc1 = __builtin_amdgcn_wmma_f32_16x16x32_f16(false, a[1], false, bfrag[1],
                                                    (short)0, acc1, false, false);
      acc0 = __builtin_amdgcn_wmma_f32_16x16x32_f16(false, a[2], false, bfrag[2],
                                                    (short)0, acc0, false, false);
      acc1 = __builtin_amdgcn_wmma_f32_16x16x32_f16(false, a[3], false, bfrag[3],
                                                    (short)0, acc1, false, false);

      // LIF update; reset reuses last step's spike flag (== (mem_prev > TH)):
      //   m = BETA*m + cur - TH*spk_prev ; spk = (m > TH) ; cnt += spk
#pragma unroll
      for (int r = 0; r < 8; ++r) {
        float cur = acc0[r] + acc1[r];
        float m   = fmaf(BETA, memv[r], cur);
        m = fmaf(spk[r], -THRESH, m);
        memv[r] = m;
        float s = (m > THRESH) ? 1.0f : 0.0f;
        spk[r] = s;
        cnt[r] += s;
      }
    }
  }

  // ---- fc2 collapsed: out[b,c] += (spike_count_tile @ W2_chunk^T) / T ----
  __syncthreads();
#pragma unroll
  for (int r = 0; r < 8; ++r)
    sS[(lane >> 4) * 8 + r][wave * 16 + (lane & 15)] = cnt[r];
  __syncthreads();

  for (int i = tid; i < BTILE * NCLASS; i += NTHREADS) {
    const int b = i / NCLASS;
    const int c = i - b * NCLASS;
    const float* w2r = W2 + (size_t)c * HIDDEN + hbase;
    float sum = 0.0f;
#pragma unroll 8
    for (int hh = 0; hh < HCHUNK; ++hh) sum += sS[b][hh] * w2r[hh];
    atomicAdd(&out[(bbase + b) * NCLASS + c], sum * (1.0f / T_STEPS));
  }
}

extern "C" void kernel_launch(void* const* d_in, const int* in_sizes, int n_in,
                              void* d_out, int out_size, void* d_ws, size_t ws_size,
                              hipStream_t stream) {
  (void)in_sizes; (void)n_in; (void)d_ws; (void)ws_size; (void)out_size;
  const float* x  = (const float*)d_in[0];
  const float* W1 = (const float*)d_in[1];
  const float* b1 = (const float*)d_in[2];
  const float* W2 = (const float*)d_in[3];
  const float* b2 = (const float*)d_in[4];
  float* out = (float*)d_out;

  snn_init_out<<<(BATCH * NCLASS + 255) / 256, 256, 0, stream>>>(b2, out);

  dim3 grid(HIDDEN / HCHUNK, BATCH / BTILE);   // (8, 32) -> 256 WGs, 1024 waves
  snn_fused<<<grid, NTHREADS, 0, stream>>>(x, W1, b1, W2, out);
}
